// Self_Attention_55456617726409
// MI455X (gfx1250) — matrix operand order repaired and checked by reading, not compiled, not run on hardware
//
#include <hip/hip_runtime.h>

// ---------------- types ----------------
typedef __attribute__((ext_vector_type(16))) __bf16        v16bf;
typedef __attribute__((ext_vector_type(8)))  float         v8f;
typedef __attribute__((ext_vector_type(4)))  unsigned int  u32x4;
typedef __attribute__((ext_vector_type(4)))  float         f32x4;

#define B_   4
#define S_   2048
#define D_   512
#define H_   8
#define DH_  64
#define KTAP 7
#define KTOT (D_*KTAP)   // 3584
#define M_   (B_*S_)     // 8192

__device__ inline unsigned short f2bf(float f) {
    unsigned int u = __float_as_uint(f);
    u += 0x7FFFu + ((u >> 16) & 1u);
    return (unsigned short)(u >> 16);
}
__device__ inline unsigned int pack2(float a, float b) {
    return (unsigned int)f2bf(a) | ((unsigned int)f2bf(b) << 16);
}

union FragU { u32x4 u[2]; v16bf v; };

__device__ inline v16bf make_frag(const unsigned short* p0, const unsigned short* p1) {
    FragU f;
    f.u[0] = *(const u32x4*)p0;
    f.u[1] = *(const u32x4*)p1;
    return f.v;
}

template<int MASK>
__device__ inline float swz_xor(float x) {
    // group-of-32 ds_swizzle: offset = xor<<10 | and_mask(0x1F)
    return __int_as_float(__builtin_amdgcn_ds_swizzle(__float_as_int(x), (MASK << 10) | 0x1F));
}

__device__ inline void wait_async0() {
#if __has_builtin(__builtin_amdgcn_s_wait_asynccnt)
    __builtin_amdgcn_s_wait_asynccnt(0);
#else
    asm volatile("s_wait_asynccnt 0" ::: "memory");
#endif
}

// Issue one async global->LDS 128-bit copy (GVS mode: saddr base + 32-bit vgpr offset).
__device__ inline void async_g2l_b128(unsigned lds_byte_off, unsigned gbl_byte_off,
                                      const void* sbase) {
    asm volatile("global_load_async_to_lds_b128 %0, %1, %2"
                 :
                 : "v"(lds_byte_off), "v"(gbl_byte_off),
                   "s"((unsigned long long)(size_t)sbase)
                 : "memory");
}

// ---------------- kernel 1: weight reshape + BN fold ----------------
__global__ void prep_kernel(const float* __restrict__ conv_w, const float* __restrict__ conv_b,
                            const float* __restrict__ bn_g, const float* __restrict__ bn_b,
                            const float* __restrict__ bn_m, const float* __restrict__ bn_v,
                            unsigned short* __restrict__ wmat,  // [7*512][512] bf16, k = t*512+ci
                            float* __restrict__ scale, float* __restrict__ bias) {
    int idx = blockIdx.x * 256 + threadIdx.x;
    if (idx < KTOT * D_) {
        int k  = idx >> 9;       // / 512
        int co = idx & 511;
        int t  = k / D_;
        int ci = k - t * D_;
        wmat[idx] = f2bf(conv_w[(co * D_ + ci) * KTAP + t]);
    }
    if (idx < D_) {
        float sc = bn_g[idx] * rsqrtf(bn_v[idx] + 1e-5f);
        scale[idx] = sc;
        bias[idx]  = (conv_b[idx] - bn_m[idx]) * sc + bn_b[idx];
    }
}

// ---------------- kernel 2: conv+BN as implicit GEMM (bf16 WMMA) ----------------
#define AS_STRIDE 40   // 32 + pad, bf16 units (80B rows, 16B aligned)
#define BS_STRIDE 40

__global__ void __launch_bounds__(256)
conv_bn_kernel(const float* __restrict__ q, const float* __restrict__ k,
               const float* __restrict__ v,
               const unsigned short* __restrict__ wmat,
               const float* __restrict__ scale, const float* __restrict__ bias,
               unsigned short* __restrict__ out) {  // [3][M_][D_] bf16
    __shared__ unsigned short As[128 * AS_STRIDE];  // [row][k]
    __shared__ unsigned short Bs[128 * BS_STRIDE];  // [n][k]  (transposed)

    const int tid  = threadIdx.x;
    const int lane = tid & 31;
    const int wid  = tid >> 5;
    const int l16  = lane & 15;
    const int hi   = lane >> 4;
    const int wm   = wid & 3;     // 4 M-groups of 32 rows
    const int wn   = wid >> 2;    // 2 N-groups of 64 cols

    const int mtile = blockIdx.x; // 0..63
    const int ntile = blockIdx.y; // 0..3
    const int tsel  = blockIdx.z; // 0..2
    const float* x = tsel == 0 ? q : (tsel == 1 ? k : v);
    unsigned short* y = out + (size_t)tsel * (size_t)M_ * D_;

    v8f acc[2][4];
#pragma unroll
    for (int i = 0; i < 2; i++)
#pragma unroll
        for (int j = 0; j < 4; j++) acc[i][j] = {};

    // A-stage mapping: 2 threads per row, 16 K each
    const int arow  = tid >> 1;
    const int ahalf = tid & 1;
    const int gm    = mtile * 128 + arow;
    const int batch = gm >> 11;
    const int srow0 = gm & (S_ - 1);
    // B-stage mapping: 16 k-pairs x 16 col-groups of 8
    const int kp = tid >> 4;
    const int cg = tid & 15;

    for (int kk = 0; kk < KTOT; kk += 32) {
        __syncthreads();
        // ---- stage A tile: gather shifted rows, f32 -> bf16 ----
        {
            int tap = kk >> 9;
            int ci0 = (kk & 511) + ahalf * 16;
            int sr  = srow0 + tap - 3;
            unsigned int packed[8];
            if (sr >= 0 && sr < S_) {
                const float* src = x + ((size_t)(batch * S_ + sr)) * D_ + ci0;
#pragma unroll
                for (int j = 0; j < 4; j++) {
                    f32x4 f = *(const f32x4*)(src + j * 4);
                    packed[j * 2 + 0] = pack2(f.x, f.y);
                    packed[j * 2 + 1] = pack2(f.z, f.w);
                }
            } else {
#pragma unroll
                for (int j = 0; j < 8; j++) packed[j] = 0u;
            }
            u32x4 a0 = { packed[0], packed[1], packed[2], packed[3] };
            u32x4 a1 = { packed[4], packed[5], packed[6], packed[7] };
            u32x4* dst = (u32x4*)&As[arow * AS_STRIDE + ahalf * 16];
            dst[0] = a0; dst[1] = a1;
        }
        // ---- stage B tile transposed into [n][k] ----
        {
            int k0 = kp * 2;
            int n0 = cg * 8;
            const unsigned short* src0 = wmat + (size_t)(kk + k0) * D_ + ntile * 128 + n0;
            if (kk + 32 < KTOT)
                __builtin_prefetch(src0 + 32 * D_, 0, 1);
            u32x4 r0 = *(const u32x4*)src0;
            u32x4 r1 = *(const u32x4*)(src0 + D_);
            const unsigned int* a = (const unsigned int*)&r0;
            const unsigned int* b = (const unsigned int*)&r1;
#pragma unroll
            for (int j = 0; j < 4; j++) {
                unsigned int lo0 = a[j] & 0xFFFFu, hi0 = a[j] >> 16;
                unsigned int lo1 = b[j] & 0xFFFFu, hi1 = b[j] >> 16;
                *(unsigned int*)&Bs[(n0 + 2 * j    ) * BS_STRIDE + k0] = lo0 | (lo1 << 16);
                *(unsigned int*)&Bs[(n0 + 2 * j + 1) * BS_STRIDE + k0] = hi0 | (hi1 << 16);
            }
        }
        __syncthreads();
        // ---- WMMA: 2x4 tiles per wave ----
        v16bf afr[2], bfr[4];
#pragma unroll
        for (int mt = 0; mt < 2; mt++) {
            int r = wm * 32 + mt * 16 + l16;
            afr[mt] = make_frag(&As[r * AS_STRIDE + hi * 8], &As[r * AS_STRIDE + 16 + hi * 8]);
        }
#pragma unroll
        for (int nt = 0; nt < 4; nt++) {
            int c = wn * 64 + nt * 16 + l16;
            bfr[nt] = make_frag(&Bs[c * BS_STRIDE + hi * 8], &Bs[c * BS_STRIDE + 16 + hi * 8]);
        }
#pragma unroll
        for (int mt = 0; mt < 2; mt++)
#pragma unroll
            for (int nt = 0; nt < 4; nt++)
                acc[mt][nt] = __builtin_amdgcn_wmma_f32_16x16x32_bf16(
                    false, afr[mt], false, bfr[nt], (short)0, acc[mt][nt], false, false);
    }

    // ---- epilogue: fused BN scale/bias, store bf16 ----
#pragma unroll
    for (int mt = 0; mt < 2; mt++) {
        int row = mtile * 128 + wm * 32 + mt * 16 + hi * 8;
#pragma unroll
        for (int nt = 0; nt < 4; nt++) {
            int col = ntile * 128 + wn * 64 + nt * 16 + l16;
            float sc = scale[col], bi = bias[col];
#pragma unroll
            for (int r = 0; r < 8; r++) {
                float val = acc[mt][nt][r] * sc + bi;
                y[(size_t)(row + r) * D_ + col] = f2bf(val);
            }
        }
    }
}

// ---------------- kernel 3: flash attention (bf16 WMMA, online softmax) ----------------
#define KS_STRIDE 72   // 64 + pad (144B rows)
#define VS_STRIDE 40   // 32 + pad (80B rows)

__global__ void __launch_bounds__(256)
attn_kernel(const unsigned short* __restrict__ qkvh,  // [3][M_][D_] bf16
            float* __restrict__ outf) {               // [M_][D_] f32 (pre-LN)
    __shared__ unsigned short Ks[32 * KS_STRIDE];     // [key][dh]
    __shared__ unsigned short Vs[64 * VS_STRIDE];     // [dh][key] (transposed)
    __shared__ unsigned short Ps[8 * 16 * 32];        // per-wave probs [16 rows][32 keys]

    const int tid  = threadIdx.x;
    const int lane = tid & 31;
    const int wid  = tid >> 5;
    const int l16  = lane & 15;
    const int hi   = lane >> 4;

    const int qtile = blockIdx.x;   // 0..15
    const int bh    = blockIdx.y;   // 0..31
    const int b     = bh >> 3;
    const int h     = bh & 7;

    const unsigned short* qb = qkvh;
    const unsigned short* kb = qkvh + (size_t)M_ * D_;
    const unsigned short* vb = qkvh + 2 * (size_t)M_ * D_;

    const int qrow = qtile * 128 + wid * 16 + l16;
    const unsigned short* qp = qb + ((size_t)(b * S_ + qrow)) * D_ + h * DH_;
    v16bf qa0 = make_frag(qp + hi * 8,      qp + 16 + hi * 8);  // dh 0..31
    v16bf qa1 = make_frag(qp + 32 + hi * 8, qp + 48 + hi * 8);  // dh 32..63

    float m[8], l[8];
    v8f o[4];
#pragma unroll
    for (int r = 0; r < 8; r++) { m[r] = -3.0e38f; l[r] = 0.f; }
#pragma unroll
    for (int d = 0; d < 4; d++) o[d] = {};

    const int ckey = tid >> 3;        // 0..31
    const int cdh  = (tid & 7) * 8;   // 0..56
    unsigned short* psh = &Ps[wid * 16 * 32];

    // uniform global base for this (b, h): async K staging uses saddr = this base
    const unsigned short* kbase_ptr = kb + ((size_t)(b * S_)) * D_ + h * DH_;
    // LDS byte offset of this thread's K-staging slot (addr[31:0] of the LDS aperture)
    const unsigned ks_lds_off = (unsigned)(size_t)&Ks[ckey * KS_STRIDE + cdh];

    for (int kb0 = 0; kb0 < S_; kb0 += 32) {
        __syncthreads();
        // ---- stage K [32][64] via async global->LDS DMA path ----
        {
            unsigned goff = (unsigned)(((kb0 + ckey) * D_ + cdh) * 2);
            async_g2l_b128(ks_lds_off, goff, kbase_ptr);
        }
        // ---- stage V transposed [64][32] (needs data transform -> VGPR path) ----
        {
            size_t roff = ((size_t)(b * S_ + kb0 + ckey)) * D_ + h * DH_ + cdh;
            u32x4 vv = *(const u32x4*)(vb + roff);
            const unsigned short* vs = (const unsigned short*)&vv;
#pragma unroll
            for (int j = 0; j < 8; j++)
                Vs[(cdh + j) * VS_STRIDE + ckey] = vs[j];
        }
        wait_async0();        // this wave's async K copy is in LDS
        __syncthreads();      // everyone's copies visible

        // ---- scores S = Q K^T (two 16x16 key tiles, K-dim 64 = 2 WMMAs each) ----
        v8f s0 = {}, s1 = {};
        {
            v16bf kf00 = make_frag(&Ks[l16 * KS_STRIDE + hi * 8],      &Ks[l16 * KS_STRIDE + 16 + hi * 8]);
            v16bf kf01 = make_frag(&Ks[l16 * KS_STRIDE + 32 + hi * 8], &Ks[l16 * KS_STRIDE + 48 + hi * 8]);
            v16bf kf10 = make_frag(&Ks[(16 + l16) * KS_STRIDE + hi * 8],      &Ks[(16 + l16) * KS_STRIDE + 16 + hi * 8]);
            v16bf kf11 = make_frag(&Ks[(16 + l16) * KS_STRIDE + 32 + hi * 8], &Ks[(16 + l16) * KS_STRIDE + 48 + hi * 8]);
            s0 = __builtin_amdgcn_wmma_f32_16x16x32_bf16(false, qa0, false, kf00, (short)0, s0, false, false);
            s0 = __builtin_amdgcn_wmma_f32_16x16x32_bf16(false, qa1, false, kf01, (short)0, s0, false, false);
            s1 = __builtin_amdgcn_wmma_f32_16x16x32_bf16(false, qa0, false, kf10, (short)0, s1, false, false);
            s1 = __builtin_amdgcn_wmma_f32_16x16x32_bf16(false, qa1, false, kf11, (short)0, s1, false, false);
        }

        // ---- online softmax (rows = VGPR idx + 8*hi, cols across 16-lane group) ----
        float p0[8], p1[8];
#pragma unroll
        for (int r = 0; r < 8; r++) {
            float a0 = s0[r] * 0.125f, a1 = s1[r] * 0.125f;
            float mx = fmaxf(a0, a1);
            mx = fmaxf(mx, swz_xor<1>(mx));
            mx = fmaxf(mx, swz_xor<2>(mx));
            mx = fmaxf(mx, swz_xor<4>(mx));
            mx = fmaxf(mx, swz_xor<8>(mx));
            float mn   = fmaxf(m[r], mx);
            float resc = __expf(m[r] - mn);
            float e0 = __expf(a0 - mn);
            float e1 = __expf(a1 - mn);
            float ls = e0 + e1;
            ls += swz_xor<1>(ls);
            ls += swz_xor<2>(ls);
            ls += swz_xor<4>(ls);
            ls += swz_xor<8>(ls);
            l[r] = l[r] * resc + ls;
            m[r] = mn;
#pragma unroll
            for (int d = 0; d < 4; d++) o[d][r] *= resc;
            p0[r] = e0; p1[r] = e1;
        }
        // ---- transpose probs C-layout -> A-layout via per-wave LDS ----
#pragma unroll
        for (int r = 0; r < 8; r++) {
            psh[(r + 8 * hi) * 32 + l16]      = f2bf(p0[r]);
            psh[(r + 8 * hi) * 32 + 16 + l16] = f2bf(p1[r]);
        }
        __syncthreads();
        v16bf pa = make_frag(&psh[l16 * 32 + hi * 8], &psh[l16 * 32 + 16 + hi * 8]);
        // ---- O += P V  (4 DH tiles) ----
#pragma unroll
        for (int d = 0; d < 4; d++) {
            int n = d * 16 + l16;
            v16bf vf = make_frag(&Vs[n * VS_STRIDE + hi * 8], &Vs[n * VS_STRIDE + 16 + hi * 8]);
            o[d] = __builtin_amdgcn_wmma_f32_16x16x32_bf16(
                false, pa, false, vf, (short)0, o[d], false, false);
        }
    }

    // ---- epilogue: normalize by l, store f32 ----
#pragma unroll
    for (int d = 0; d < 4; d++) {
        int col = h * DH_ + d * 16 + l16;
#pragma unroll
        for (int r = 0; r < 8; r++) {
            int row = qtile * 128 + wid * 16 + r + 8 * hi;
            outf[(size_t)(b * S_ + row) * D_ + col] = o[d][r] / l[r];
        }
    }
}

// ---------------- kernel 4: LayerNorm (one wave per row) ----------------
__global__ void __launch_bounds__(256)
ln_kernel(const float* __restrict__ x, const float* __restrict__ g,
          const float* __restrict__ be, float* __restrict__ out) {
    int wid  = threadIdx.x >> 5;
    int lane = threadIdx.x & 31;
    int row  = blockIdx.x * 8 + wid;
    const float* xr = x + (size_t)row * D_;
    f32x4 v0[4];
    float s = 0.f, ss = 0.f;
#pragma unroll
    for (int j = 0; j < 4; j++) {
        v0[j] = *(const f32x4*)(xr + lane * 16 + j * 4);
#pragma unroll
        for (int e = 0; e < 4; e++) { float f = v0[j][e]; s += f; ss += f * f; }
    }
    s  += swz_xor<1>(s);  s  += swz_xor<2>(s);  s  += swz_xor<4>(s);
    s  += swz_xor<8>(s);  s  += swz_xor<16>(s);
    ss += swz_xor<1>(ss); ss += swz_xor<2>(ss); ss += swz_xor<4>(ss);
    ss += swz_xor<8>(ss); ss += swz_xor<16>(ss);
    float mu  = s * (1.f / 512.f);
    float var = ss * (1.f / 512.f) - mu * mu;
    float inv = rsqrtf(var + 1e-5f);
    float* orow = out + (size_t)row * D_;
#pragma unroll
    for (int j = 0; j < 4; j++) {
#pragma unroll
        for (int e = 0; e < 4; e++) {
            int c = lane * 16 + j * 4 + e;
            v0[j][e] = (v0[j][e] - mu) * inv * g[c] + be[c];
        }
        *(f32x4*)(orow + lane * 16 + j * 4) = v0[j];
    }
}

// ---------------- host launch ----------------
extern "C" void kernel_launch(void* const* d_in, const int* in_sizes, int n_in,
                              void* d_out, int out_size, void* d_ws, size_t ws_size,
                              hipStream_t stream) {
    const float* q      = (const float*)d_in[0];
    const float* k      = (const float*)d_in[1];
    const float* v      = (const float*)d_in[2];
    const float* conv_w = (const float*)d_in[3];
    const float* conv_b = (const float*)d_in[4];
    const float* bn_g   = (const float*)d_in[5];
    const float* bn_b   = (const float*)d_in[6];
    const float* bn_m   = (const float*)d_in[7];
    const float* bn_v   = (const float*)d_in[8];
    const float* ln_g   = (const float*)d_in[9];
    const float* ln_b   = (const float*)d_in[10];

    char* ws = (char*)d_ws;
    unsigned short* wmat  = (unsigned short*)ws;                       // 3584*512 bf16 = 3,670,016 B
    float*          scale = (float*)(ws + 3670016);                    // 512 f32
    float*          bias  = (float*)(ws + 3670016 + 2048);             // 512 f32
    unsigned short* qkvh  = (unsigned short*)(ws + 3674112);           // 3*8192*512 bf16 = 25,165,824 B
    float*          attn  = (float*)(ws + 3674112 + 25165824);         // 8192*512 f32  = 16,777,216 B

    prep_kernel<<<7168, 256, 0, stream>>>(conv_w, conv_b, bn_g, bn_b, bn_m, bn_v,
                                          wmat, scale, bias);
    conv_bn_kernel<<<dim3(64, 4, 3), 256, 0, stream>>>(q, k, v, wmat, scale, bias, qkvh);
    attn_kernel<<<dim3(16, 32), 256, 0, stream>>>(qkvh, attn);
    ln_kernel<<<1024, 256, 0, stream>>>(attn, ln_g, ln_b, (float*)d_out);
}
